// T_update_all_83434034692271
// MI455X (gfx1250) — matrix operand order repaired and checked by reading, not compiled, not run on hardware
//
#include <hip/hip_runtime.h>
#include <hip/hip_bf16.h>

#define DEVINL __device__ __forceinline__

typedef __bf16 bf16_t;
typedef __attribute__((ext_vector_type(16))) __bf16 v16bf;
typedef __attribute__((ext_vector_type(8)))  __bf16 v8bf;
typedef __attribute__((ext_vector_type(8)))  float   v8f;

// Problem constants (reference: B=2, L1=L2=1024, D=512, H=8, DK=64)
static constexpr int CB  = 2;
static constexpr int CH  = 8;
static constexpr int CL  = 1024;
static constexpr int CD  = 512;
static constexpr int CDK = 64;

DEVINL v16bf cat8(v8bf lo, v8bf hi) {
  return __builtin_shufflevector(lo, hi, 0,1,2,3,4,5,6,7,8,9,10,11,12,13,14,15);
}

// A fragment (16x32 bf16). lane = M row (lane%16); K = 16*(lane/16)+[0..15]
// contiguous, matching the B fragment order.
DEVINL v16bf frag_a_bf16(const bf16_t* __restrict__ p, int ld, int lane) {
  const int r = lane & 15, h2 = lane >> 4;
  const bf16_t* q = p + (size_t)r * ld + 16 * h2;
  v8bf lo = *(const v8bf*)(q);
  v8bf hi = *(const v8bf*)(q + 8);
  return cat8(lo, hi);
}

DEVINL v16bf frag_a_f32(const float* __restrict__ p, int ld, int lane) {
  const int r = lane & 15, h2 = lane >> 4;
  const float* q = p + (size_t)r * ld + 16 * h2;
  v16bf o;
#pragma unroll
  for (int i = 0; i < 8; ++i) o[i] = (bf16_t)q[i];
#pragma unroll
  for (int i = 0; i < 8; ++i) o[8 + i] = (bf16_t)q[8 + i];
  return o;
}

// B fragment (32x16 bf16). lane = N col (lane%16); K = 16*(lane/16)+[0..15]
// contiguous, matching the A fragment order.
DEVINL v16bf frag_b_bf16(const bf16_t* __restrict__ p, int ld, int lane) {
  const int n = lane & 15, h2 = lane >> 4;
  const bf16_t* q = p + (size_t)n * ld + 16 * h2;
  v8bf lo = *(const v8bf*)(q);
  v8bf hi = *(const v8bf*)(q + 8);
  return cat8(lo, hi);
}

DEVINL v16bf frag_b_f32(const float* __restrict__ p, int ld, int lane) {
  const int n = lane & 15, h2 = lane >> 4;
  const float* q = p + (size_t)n * ld + 16 * h2;
  v16bf o;
#pragma unroll
  for (int i = 0; i < 16; ++i) o[i] = (bf16_t)q[i];
  return o;
}

DEVINL v8f wmma_bf16(v16bf a, v16bf b, v8f c) {
  return __builtin_amdgcn_wmma_f32_16x16x32_bf16(false, a, false, b, (short)0, c,
                                                 false, false);
}

DEVINL float red_max16(float v) {
  v = fmaxf(v, __shfl_xor(v, 1, 32));
  v = fmaxf(v, __shfl_xor(v, 2, 32));
  v = fmaxf(v, __shfl_xor(v, 4, 32));
  v = fmaxf(v, __shfl_xor(v, 8, 32));
  return v;
}
DEVINL float red_sum16(float v) {
  v += __shfl_xor(v, 1, 32);
  v += __shfl_xor(v, 2, 32);
  v += __shfl_xor(v, 4, 32);
  v += __shfl_xor(v, 8, 32);
  return v;
}

// ---------------------------------------------------------------------------
// Kernel 1: fused Q/K/V projection (blockIdx.z selects matrix).
// Y = X @ W^T + bias, M=2048, N=512, K=512, fp32 in -> bf16 out, head-split.
// q,k stored [BH, L, 64]; v stored transposed [BH, 64, L] for the aw@v GEMM.
// ---------------------------------------------------------------------------
__global__ __launch_bounds__(256) void proj_qkv_kernel(
    const float* __restrict__ query, const float* __restrict__ key,
    const float* __restrict__ value,
    const float* __restrict__ Wq, const float* __restrict__ bq,
    const float* __restrict__ Wk, const float* __restrict__ bk,
    const float* __restrict__ Wv, const float* __restrict__ bv,
    bf16_t* __restrict__ q_ws, bf16_t* __restrict__ k_ws,
    bf16_t* __restrict__ vT_ws)
{
  const int lane = threadIdx.x & 31, wave = threadIdx.x >> 5;
  const int mw = wave & 1, nw = wave >> 1;
  const int mat = blockIdx.z;
  const float* X = (mat == 0) ? query : (mat == 1) ? key : value;
  const float* W = (mat == 0) ? Wq : (mat == 1) ? Wk : Wv;
  const float* bias = (mat == 0) ? bq : (mat == 1) ? bk : bv;
  const int m0 = blockIdx.x * 64 + mw * 32;
  const int n0 = blockIdx.y * 256 + nw * 64;

  v8f acc[2][4];
#pragma unroll
  for (int i = 0; i < 2; ++i)
#pragma unroll
    for (int j = 0; j < 4; ++j) acc[i][j] = v8f{};

  for (int k0 = 0; k0 < CD; k0 += 32) {
    v16bf a[2], bfr[4];
#pragma unroll
    for (int mt = 0; mt < 2; ++mt)
      a[mt] = frag_a_f32(X + (size_t)(m0 + mt * 16) * CD + k0, CD, lane);
#pragma unroll
    for (int nt = 0; nt < 4; ++nt)
      bfr[nt] = frag_b_f32(W + (size_t)(n0 + nt * 16) * CD + k0, CD, lane);
#pragma unroll
    for (int mt = 0; mt < 2; ++mt)
#pragma unroll
      for (int nt = 0; nt < 4; ++nt)
        acc[mt][nt] = wmma_bf16(a[mt], bfr[nt], acc[mt][nt]);
  }

  const int r16 = lane & 15, h2 = lane >> 4;
#pragma unroll
  for (int mt = 0; mt < 2; ++mt)
#pragma unroll
    for (int nt = 0; nt < 4; ++nt) {
      const int ncol = n0 + nt * 16 + r16;
      const int hh = ncol >> 6, dk = ncol & 63;
      const float bsv = bias[ncol];
#pragma unroll
      for (int r = 0; r < 8; ++r) {
        const int mg = m0 + mt * 16 + r + 8 * h2;
        const int bb = mg >> 10, l = mg & (CL - 1);
        const bf16_t val = (bf16_t)(acc[mt][nt][r] + bsv);
        const size_t nat = ((size_t)(bb * CH + hh) * CL + l) * CDK + dk;
        if (mat == 0)      q_ws[nat] = val;
        else if (mat == 1) k_ws[nat] = val;
        else vT_ws[((size_t)(bb * CH + hh) * CDK + dk) * CL + l] = val;
      }
    }
}

// ---------------------------------------------------------------------------
// Kernel 2: ebias[b,h,key] = sum_d v[b,h,key,d]*we[d] + be   (from vT, coalesced)
// ---------------------------------------------------------------------------
__global__ void ebias_kernel(const bf16_t* __restrict__ vT,
                             const float* __restrict__ we,
                             const float* __restrict__ be,
                             float* __restrict__ eb)
{
  const int i = blockIdx.x * blockDim.x + threadIdx.x;  // [0, BH*L)
  const int bh = i >> 10, keyi = i & (CL - 1);
  float s = be[0];
  const bf16_t* base = vT + (size_t)bh * CDK * CL + keyi;
#pragma unroll 8
  for (int d = 0; d < CDK; ++d) s += (float)base[(size_t)d * CL] * we[d];
  eb[i] = s;
}

// ---------------------------------------------------------------------------
// Kernel 3: fused attention. One wave = 16 query rows of one (b,h).
// scores -> attn1 = scores*(ebias*edge) -> edge_new store -> masked online
// softmax -> O += P @ v (via per-wave LDS repack of P into A layout).
// ---------------------------------------------------------------------------
__global__ __launch_bounds__(256) void attn_kernel(
    const bf16_t* __restrict__ q_ws, const bf16_t* __restrict__ k_ws,
    const bf16_t* __restrict__ vT_ws, const float* __restrict__ ebias,
    const float* __restrict__ emo_mask, const float* __restrict__ edge,
    const float* __restrict__ w1p, const float* __restrict__ b1p,
    float* __restrict__ edge_new, bf16_t* __restrict__ x0)
{
  __shared__ __align__(16) bf16_t plds[8][16][72];  // per-wave P staging

  const int lane = threadIdx.x & 31, w = threadIdx.x >> 5;
  const int r16 = lane & 15, h2 = lane >> 4;
  const int bh = blockIdx.x >> 3;
  const int q0 = (blockIdx.x & 7) * 128 + w * 16;
  const int b = bh >> 3, h = bh & 7;
  const float w1 = w1p[0], b1 = b1p[0];

  const bf16_t* qb   = q_ws + ((size_t)bh * CL + q0) * CDK;
  const bf16_t* kb_  = k_ws + (size_t)bh * CL * CDK;
  const bf16_t* vtb  = vT_ws + (size_t)bh * CDK * CL;
  const float*  edg  = edge     + (size_t)bh * CL * CL;
  const float*  msk  = emo_mask + (size_t)bh * CL * CL;
  float*        eno  = edge_new + (size_t)bh * CL * CL;
  const float*  ebv  = ebias + (size_t)bh * CL;

  const v16bf qa0 = frag_a_bf16(qb, CDK, lane);        // dk 0..31
  const v16bf qa1 = frag_a_bf16(qb + 32, CDK, lane);   // dk 32..63

  v8f o[4];
#pragma unroll
  for (int t = 0; t < 4; ++t) o[t] = v8f{};
  float mrow[8], lrow[8];
#pragma unroll
  for (int r = 0; r < 8; ++r) { mrow[r] = -3.0e38f; lrow[r] = 0.0f; }

  for (int key0 = 0; key0 < CL; key0 += 32) {
    v8f s0 = v8f{}, s1 = v8f{};
    {
      v16bf kb;
      kb = frag_b_bf16(kb_ + (size_t)(key0     ) * CDK +  0, CDK, lane);
      s0 = wmma_bf16(qa0, kb, s0);
      kb = frag_b_bf16(kb_ + (size_t)(key0     ) * CDK + 32, CDK, lane);
      s0 = wmma_bf16(qa1, kb, s0);
      kb = frag_b_bf16(kb_ + (size_t)(key0 + 16) * CDK +  0, CDK, lane);
      s1 = wmma_bf16(qa0, kb, s1);
      kb = frag_b_bf16(kb_ + (size_t)(key0 + 16) * CDK + 32, CDK, lane);
      s1 = wmma_bf16(qa1, kb, s1);
    }

    const float eb0 = ebv[key0 + r16];
    const float eb1 = ebv[key0 + 16 + r16];

    float p0[8], p1[8], nm[8];
#pragma unroll
    for (int r = 0; r < 8; ++r) {
      const int qm = q0 + r + 8 * h2;
      const size_t ei = (size_t)qm * CL + key0 + r16;
      const float ed0 = edg[ei], ed1 = edg[ei + 16];
      const float mk0 = msk[ei], mk1 = msk[ei + 16];
      float a0 = s0[r] * (eb0 * ed0);
      float a1 = s1[r] * (eb1 * ed1);
      eno[ei]      = (ed0 + a0) * w1 + b1;   // edge_new = (edge+attn1)*w1+b1
      eno[ei + 16] = (ed1 + a1) * w1 + b1;
      a0 -= mk0 * 1e30f;
      a1 -= mk1 * 1e30f;
      nm[r] = fmaxf(mrow[r], red_max16(fmaxf(a0, a1)));
      p0[r] = a0; p1[r] = a1;
    }

#pragma unroll
    for (int r = 0; r < 8; ++r) {
      const float mn = nm[r];
      const float sc = __expf(mrow[r] - mn);
      mrow[r] = mn;
      const float e0 = __expf(p0[r] - mn);
      const float e1 = __expf(p1[r] - mn);
      lrow[r] = lrow[r] * sc + red_sum16(e0 + e1);
#pragma unroll
      for (int t = 0; t < 4; ++t) o[t][r] *= sc;
      const int ml = r + 8 * h2;
      plds[w][ml][r16]      = (bf16_t)e0;   // C layout -> LDS row-major
      plds[w][ml][16 + r16] = (bf16_t)e1;
    }
    asm volatile("s_wait_dscnt 0" ::: "memory");

    const v16bf pa = frag_a_bf16(&plds[w][0][0], 72, lane);
#pragma unroll
    for (int t = 0; t < 4; ++t) {
      const v16bf vb = frag_b_bf16(vtb + (size_t)(t * 16) * CL + key0, CL, lane);
      o[t] = wmma_bf16(pa, vb, o[t]);
    }
  }

#pragma unroll
  for (int r = 0; r < 8; ++r) {
    const float inv = 1.0f / lrow[r];
    const int qm = q0 + r + 8 * h2;
    bf16_t* xr = x0 + ((size_t)b * CL + qm) * CD + h * CDK + r16;
#pragma unroll
    for (int t = 0; t < 4; ++t) xr[t * 16] = (bf16_t)(o[t][r] * inv);
  }
}

// ---------------------------------------------------------------------------
// Kernel 4: GraphLayer GEMM  Y = relu?(X @ W^T). M=2048 fixed.
// TRIPLE==1: write fp32 to d_out three times (tuple (out,out,out,...)).
// ---------------------------------------------------------------------------
template <int RELU, int TRIPLE>
__global__ __launch_bounds__(256) void gemm_kernel(
    const bf16_t* __restrict__ X, const float* __restrict__ W,
    bf16_t* __restrict__ Yb, float* __restrict__ Yf, int N, int K)
{
  const int lane = threadIdx.x & 31, wave = threadIdx.x >> 5;
  const int mw = wave & 1, nw = wave >> 1;
  const int m0 = blockIdx.x * 64 + mw * 32;
  const int n0 = blockIdx.y * 256 + nw * 64;

  v8f acc[2][4];
#pragma unroll
  for (int i = 0; i < 2; ++i)
#pragma unroll
    for (int j = 0; j < 4; ++j) acc[i][j] = v8f{};

  for (int k0 = 0; k0 < K; k0 += 32) {
    if (k0 + 32 < K) {
      __builtin_prefetch(X + (size_t)m0 * K + k0 + 32, 0, 0);
      __builtin_prefetch(W + (size_t)n0 * K + k0 + 32, 0, 0);
    }
    v16bf a[2], bfr[4];
#pragma unroll
    for (int mt = 0; mt < 2; ++mt)
      a[mt] = frag_a_bf16(X + (size_t)(m0 + mt * 16) * K + k0, K, lane);
#pragma unroll
    for (int nt = 0; nt < 4; ++nt)
      bfr[nt] = frag_b_f32(W + (size_t)(n0 + nt * 16) * K + k0, K, lane);
#pragma unroll
    for (int mt = 0; mt < 2; ++mt)
#pragma unroll
      for (int nt = 0; nt < 4; ++nt)
        acc[mt][nt] = wmma_bf16(a[mt], bfr[nt], acc[mt][nt]);
  }

  const int r16 = lane & 15, h2 = lane >> 4;
#pragma unroll
  for (int mt = 0; mt < 2; ++mt)
#pragma unroll
    for (int nt = 0; nt < 4; ++nt) {
      const int ncol = n0 + nt * 16 + r16;
#pragma unroll
      for (int r = 0; r < 8; ++r) {
        const int mg = m0 + mt * 16 + r + 8 * h2;
        float val = acc[mt][nt][r];
        if (RELU) val = fmaxf(val, 0.0f);
        if (TRIPLE) {
          const size_t idx = (size_t)mg * CD + ncol;
          Yf[idx] = val;
          Yf[idx + (size_t)CB * CL * CD] = val;
          Yf[idx + (size_t)2 * CB * CL * CD] = val;
        } else {
          Yb[(size_t)mg * N + ncol] = (bf16_t)val;
        }
      }
    }
}

// ---------------------------------------------------------------------------
extern "C" void kernel_launch(void* const* d_in, const int* in_sizes, int n_in,
                              void* d_out, int out_size, void* d_ws, size_t ws_size,
                              hipStream_t stream)
{
  const float* query    = (const float*)d_in[0];
  const float* key      = (const float*)d_in[1];
  const float* value    = (const float*)d_in[2];
  const float* emo_mask = (const float*)d_in[3];
  const float* edge     = (const float*)d_in[4];
  // d_in[5] = emo_emb (unused by reference)
  const float* Wq = (const float*)d_in[6];
  const float* bq = (const float*)d_in[7];
  const float* Wk = (const float*)d_in[8];
  const float* bk = (const float*)d_in[9];
  const float* Wv = (const float*)d_in[10];
  const float* bv = (const float*)d_in[11];
  const float* we = (const float*)d_in[12];
  const float* be = (const float*)d_in[13];
  const float* w1 = (const float*)d_in[14];
  const float* b1 = (const float*)d_in[15];
  const float* G0W1 = (const float*)d_in[16];
  const float* G0W2 = (const float*)d_in[17];
  const float* G1W1 = (const float*)d_in[18];
  const float* G1W2 = (const float*)d_in[19];
  (void)in_sizes; (void)n_in; (void)out_size; (void)ws_size;

  char* ws = (char*)d_ws;
  bf16_t* q_ws  = (bf16_t*)(ws + ((size_t)0  << 20));  // 2 MB
  bf16_t* k_ws  = (bf16_t*)(ws + ((size_t)2  << 20));  // 2 MB
  bf16_t* vT_ws = (bf16_t*)(ws + ((size_t)4  << 20));  // 2 MB
  float*  eb_ws = (float*) (ws + ((size_t)6  << 20));  // 64 KB
  bf16_t* x0    = (bf16_t*)(ws + ((size_t)7  << 20));  // 2 MB
  bf16_t* x1    = (bf16_t*)(ws + ((size_t)9  << 20));  // 4 MB
  bf16_t* x2    = (bf16_t*)(ws + ((size_t)13 << 20));  // 2 MB

  float* out = (float*)d_out;
  float* edge_new = out + (size_t)3 * CB * CL * CD;  // after 3 copies of out

  proj_qkv_kernel<<<dim3(32, 2, 3), 256, 0, stream>>>(
      query, key, value, Wq, bq, Wk, bk, Wv, bv, q_ws, k_ws, vT_ws);

  ebias_kernel<<<dim3((CB * CH * CL) / 256), 256, 0, stream>>>(vT_ws, we, be, eb_ws);

  attn_kernel<<<dim3(CB * CH * 8), 256, 0, stream>>>(
      q_ws, k_ws, vT_ws, eb_ws, emo_mask, edge, w1, b1, edge_new, x0);

  gemm_kernel<1, 0><<<dim3(32, 4), 256, 0, stream>>>(x0, G0W1, x1, nullptr, 1024, 512);
  gemm_kernel<0, 0><<<dim3(32, 2), 256, 0, stream>>>(x1, G0W2, x2, nullptr, 512, 1024);
  gemm_kernel<1, 0><<<dim3(32, 4), 256, 0, stream>>>(x2, G1W1, x1, nullptr, 1024, 512);
  gemm_kernel<0, 1><<<dim3(32, 2), 256, 0, stream>>>(x1, G1W2, nullptr, out, 512, 1024);
}
